// MultiHeadAttention_40166534152797
// MI455X (gfx1250) — matrix operand order repaired and checked by reading, not compiled, not run on hardware
//
#include <hip/hip_runtime.h>
#include <hip/hip_bf16.h>

// MHA forward for MI455X (gfx1250): bf16 WMMA (v_wmma_f32_16x16x32_bf16) with
// f32 accumulation, TDM (tensor_load_to_lds) double-buffered staging in the
// fused ReLU-attention kernel, LDS-staged GEMMs with global prefetch.

typedef __attribute__((ext_vector_type(16))) __bf16        v16bf;
typedef __attribute__((ext_vector_type(8)))  float         v8f;
typedef __attribute__((ext_vector_type(4)))  unsigned int  v4u;
typedef __attribute__((ext_vector_type(8)))  int           v8i;
typedef __attribute__((ext_vector_type(4)))  int           v4i;

#define D_MODEL 1024
#define SEQ     2048
#define BATCH   2
#define NHEAD   8
#define HDIM    128

// ---- WMMA fragment loaders (ISA 7.12.2 layouts, wave32) --------------------
// A-matrix 16x32 bf16: lane holds row M=lane%16; with klo=(lane/16)*8,
// v0..v3 = K {klo..klo+7}, v4..v7 = K {klo+16..klo+23}. K-pairs contiguous.
__device__ __forceinline__ v16bf frag_a(const __bf16* base, int stride, int lane) {
  const __bf16* p = base + (lane & 15) * stride + ((lane >> 4) << 3);
  v16bf f;
  unsigned int* fi = reinterpret_cast<unsigned int*>(&f);
#pragma unroll
  for (int i = 0; i < 4; ++i) fi[i]     = *reinterpret_cast<const unsigned int*>(p + 2 * i);
#pragma unroll
  for (int i = 0; i < 4; ++i) fi[4 + i] = *reinterpret_cast<const unsigned int*>(p + 16 + 2 * i);
  return f;
}

// B-matrix 32x16 bf16 (column-major storage: base[n*stride + k]):
// lane holds column N=lane%16; with klo=(lane/16)*8, K {klo..klo+7, klo+16..klo+23}.
__device__ __forceinline__ v16bf frag_b(const __bf16* base, int stride, int lane) {
  const __bf16* p = base + (lane & 15) * stride + ((lane >> 4) << 3);
  v16bf f;
  unsigned int* fi = reinterpret_cast<unsigned int*>(&f);
#pragma unroll
  for (int i = 0; i < 4; ++i) fi[i]     = *reinterpret_cast<const unsigned int*>(p + 2 * i);
#pragma unroll
  for (int i = 0; i < 4; ++i) fi[4 + i] = *reinterpret_cast<const unsigned int*>(p + 16 + 2 * i);
  return f;
}

// ---- TDM: 2-D tile DMA global -> LDS (ISA ch. 8 descriptor) ----------------
// tile_d0 elements per row (contiguous, bf16), tile_d1 rows, row stride
// `stride0` elements. LDS rows padded: pad_interval (log2 of 8-byte units per
// row) / pad_amount (code N -> (N+1) dwords appended per row).
__device__ __forceinline__ void tdm_load_2d(unsigned int lds_addr,
                                            const void* gaddr,
                                            unsigned int tile_d0,
                                            unsigned int tile_d1,
                                            unsigned long long stride0,
                                            unsigned int pad_interval,
                                            unsigned int pad_amount) {
  const unsigned long long ga = (unsigned long long)(size_t)gaddr;
  v4u g0;
  g0[0] = 1u;                                              // count=1, user load
  g0[1] = lds_addr;                                        // LDS byte address
  g0[2] = (unsigned int)ga;                                // global_addr[31:0]
  g0[3] = (unsigned int)((ga >> 32) & 0x1FFFFFFu) | (2u << 30);  // [56:32]|type=2

  const unsigned int td0 = 1u << 20;                       // generous dims: no OOB
  const unsigned int td1 = 1u << 20;
  const unsigned int s0lo = (unsigned int)(stride0 & 0xFFFFFFFFull);
  const unsigned int s0hi = (unsigned int)((stride0 >> 32) & 0xFFFFull);
  v8i g1;
  g1[0] = (int)((1u << 16)                                 // data_size = 2 bytes
              | (1u << 20)                                 // pad_enable
              | (pad_interval << 22) | (pad_amount << 25));
  g1[1] = (int)(td0 << 16);                                // [63:48]=dim0 lo16
  g1[2] = (int)((td0 >> 16) | (td1 << 16));                // dim0 hi | dim1 lo
  g1[3] = (int)((td1 >> 16) | (tile_d0 << 16));            // dim1 hi | tile0
  g1[4] = (int)(tile_d1 & 0xFFFFu);                        // tile1 | tile2=0
  g1[5] = (int)s0lo;                                       // dim0_stride[31:0]
  g1[6] = (int)s0hi;                                       // stride hi | dim1_stride=0
  g1[7] = 0;

  const v4i z4 = {0, 0, 0, 0};
#if defined(__clang_major__) && (__clang_major__ >= 23)
  const v8i z8 = {0, 0, 0, 0, 0, 0, 0, 0};
  __builtin_amdgcn_tensor_load_to_lds(g0, g1, z4, z4, z8, 0);
#else
  __builtin_amdgcn_tensor_load_to_lds(g0, g1, z4, z4, 0);
#endif
}

// ---- GEMM: Out[M,N] = A[M,K] @ B[K,N] + bias -------------------------------
// Block tile 128x128x32, 256 threads = 8 waves (2x4), wave tile 64x32.
// OUT_MODE: 0 = bf16 row-major, 1 = f32 row-major, 2 = bf16 scattered to
// per-head-transposed layout [B,H,HDIM,SEQ] (for the V projection).
template <bool A_IS_F32, int OUT_MODE>
__global__ __launch_bounds__(256) void gemm_bias_kernel(
    const void* __restrict__ Ap, const float* __restrict__ Bp,
    const float* __restrict__ bias, void* __restrict__ Outp,
    int M, int N, int K) {
  __shared__ __bf16 As[128][40];   // [m][k], stride 40 (pad)
  __shared__ __bf16 Bs[128][40];   // [n][k] (transposed), stride 40

  const int tid  = threadIdx.x;
  const int lane = tid & 31;
  const int wave = tid >> 5;
  const int wm   = wave >> 2;          // 0..1
  const int wn   = wave & 3;           // 0..3
  const int m0   = blockIdx.y * 128;
  const int n0   = blockIdx.x * 128;

  const v8f vzero = {0, 0, 0, 0, 0, 0, 0, 0};
  v8f acc[4][2];
#pragma unroll
  for (int i = 0; i < 4; ++i)
#pragma unroll
    for (int j = 0; j < 2; ++j) acc[i][j] = vzero;

  const int arow = tid >> 3;           // 0..31
  const int acol = (tid & 7) * 4;      // 0..28
  const int brow = tid >> 5;           // 0..7
  const int bcol = lane * 4;           // 0..124

  for (int k0 = 0; k0 < K; k0 += 32) {
    // Prefetch next k-tile while we stage/compute this one.
    if (k0 + 32 < K) {
      if (A_IS_F32)
        __builtin_prefetch(reinterpret_cast<const float*>(Ap) +
                               (size_t)(m0 + arow) * K + (k0 + 32) + acol, 0, 3);
      __builtin_prefetch(Bp + (size_t)(k0 + 32 + brow) * N + n0 + bcol, 0, 3);
    }
    __syncthreads();
    // Stage A tile (128x32), f32->bf16 on the fly when needed.
#pragma unroll
    for (int p = 0; p < 4; ++p) {
      const int r = p * 32 + arow;
      __bf16* dst = &As[r][acol];
      if (A_IS_F32) {
        const float4 v = *reinterpret_cast<const float4*>(
            reinterpret_cast<const float*>(Ap) + (size_t)(m0 + r) * K + k0 + acol);
        dst[0] = (__bf16)v.x; dst[1] = (__bf16)v.y;
        dst[2] = (__bf16)v.z; dst[3] = (__bf16)v.w;
      } else {
        *reinterpret_cast<uint2*>(dst) = *reinterpret_cast<const uint2*>(
            reinterpret_cast<const __bf16*>(Ap) + (size_t)(m0 + r) * K + k0 + acol);
      }
    }
    // Stage B tile (32x128) transposed into Bs[n][k].
#pragma unroll
    for (int p = 0; p < 4; ++p) {
      const int kk = p * 8 + brow;
      const float4 v = *reinterpret_cast<const float4*>(
          Bp + (size_t)(k0 + kk) * N + n0 + bcol);
      Bs[bcol + 0][kk] = (__bf16)v.x;
      Bs[bcol + 1][kk] = (__bf16)v.y;
      Bs[bcol + 2][kk] = (__bf16)v.z;
      Bs[bcol + 3][kk] = (__bf16)v.w;
    }
    __syncthreads();

    v16bf a[4], b[2];
#pragma unroll
    for (int mf = 0; mf < 4; ++mf) a[mf] = frag_a(&As[wm * 64 + mf * 16][0], 40, lane);
#pragma unroll
    for (int nf = 0; nf < 2; ++nf) b[nf] = frag_b(&Bs[wn * 32 + nf * 16][0], 40, lane);
#pragma unroll
    for (int mf = 0; mf < 4; ++mf)
#pragma unroll
      for (int nf = 0; nf < 2; ++nf)
        acc[mf][nf] = __builtin_amdgcn_wmma_f32_16x16x32_bf16(
            false, a[mf], false, b[nf], (short)0, acc[mf][nf], false, false);
  }

  // Epilogue: C/D layout — VGPR i holds M = i + (lane/16)*8, N = lane%16.
  const int crow = (lane >> 4) * 8;
  const int ccol = lane & 15;
#pragma unroll
  for (int mf = 0; mf < 4; ++mf) {
#pragma unroll
    for (int nf = 0; nf < 2; ++nf) {
      const int col = n0 + wn * 32 + nf * 16 + ccol;
      const float bv = bias[col];
      if (OUT_MODE == 2) {
        // V projection: rows are tokens (b,key); this lane's 8 VGPR entries are
        // 8 consecutive keys -> one packed 16B store into [B,H,HDIM,SEQ].
        const int row0 = m0 + wm * 64 + mf * 16 + crow;    // global token row
        const int bidx = row0 / SEQ;
        const int key  = row0 % SEQ;
        const int hh   = col >> 7;                         // head
        const int hd   = col & (HDIM - 1);
        union { uint4 u; __bf16 e[8]; } pk;
#pragma unroll
        for (int i = 0; i < 8; ++i) pk.e[i] = (__bf16)(acc[mf][nf][i] + bv);
        *reinterpret_cast<uint4*>(
            reinterpret_cast<__bf16*>(Outp) +
            ((size_t)(bidx * NHEAD + hh) * HDIM + hd) * SEQ + key) = pk.u;
      } else {
#pragma unroll
        for (int i = 0; i < 8; ++i) {
          const int row = m0 + wm * 64 + mf * 16 + crow + i;
          const float v = acc[mf][nf][i] + bv;
          if (OUT_MODE == 1)
            reinterpret_cast<float*>(Outp)[(size_t)row * N + col] = v;
          else
            reinterpret_cast<__bf16*>(Outp)[(size_t)row * N + col] = (__bf16)v;
        }
      }
    }
  }
}

// ---- Fused ReLU attention ---------------------------------------------------
// One block per (batch, head, 64-query tile); 128 threads = 4 waves.
// K tiles ([key][hd]) and pre-transposed V tiles ([hd][key]) are DMA'd into
// double-buffered LDS by the Tensor Data Mover (wave 0 issues, TENSORcnt
// tracked), overlapping DMA of tile kt+1 with WMMA on tile kt.
__global__ __launch_bounds__(128) void attn_kernel(
    const __bf16* __restrict__ Q, const __bf16* __restrict__ Kmat,
    const __bf16* __restrict__ Vtg, float* __restrict__ attn_w,
    __bf16* __restrict__ ctx) {
  __shared__ __bf16 Ks[2][64][136];   // [key][hd], stride 136 (TDM pad) 34.8KB
  __shared__ __bf16 Vt[2][128][72];   // [hd][key], stride 72  (TDM pad) 36.9KB
  __shared__ __bf16 Ps[64][72];       // [q][key]                        9.2KB

  const int tid  = threadIdx.x;
  const int lane = tid & 31;
  const int wave = tid >> 5;       // 0..3
  const int b    = blockIdx.z;
  const int h    = blockIdx.y;
  const int q0   = blockIdx.x * 64;
  const float scale = 0.088388347648318447f;   // 1/sqrt(128)

  // Preload this wave's Q A-fragments (16 rows x 128 hd = 4 frags) from global.
  v16bf qa[4];
  {
    const __bf16* qp = Q +
        (size_t)(b * SEQ + q0 + wave * 16 + (lane & 15)) * D_MODEL +
        h * HDIM + ((lane >> 4) << 3);
#pragma unroll
    for (int kf = 0; kf < 4; ++kf) {
      unsigned int* fi = reinterpret_cast<unsigned int*>(&qa[kf]);
      const __bf16* p = qp + kf * 32;
#pragma unroll
      for (int i = 0; i < 4; ++i) fi[i]     = *reinterpret_cast<const unsigned int*>(p + 2 * i);
#pragma unroll
      for (int i = 0; i < 4; ++i) fi[4 + i] = *reinterpret_cast<const unsigned int*>(p + 16 + 2 * i);
    }
  }

  const v8f vzero = {0, 0, 0, 0, 0, 0, 0, 0};
  v8f cacc[8];
#pragma unroll
  for (int i = 0; i < 8; ++i) cacc[i] = vzero;

  const int crow = (lane >> 4) * 8;
  const int ccol = lane & 15;
  const int NT   = SEQ / 64;

  // TDM issue for key-tile kt into LDS buffer `buf` (wave 0 only, once).
  auto issue_tile = [&](int kt, int buf) {
    const __bf16* kg = Kmat + (size_t)(b * SEQ + kt * 64) * D_MODEL + h * HDIM;
    tdm_load_2d((unsigned int)(size_t)&Ks[buf][0][0], kg,
                /*tile_d0=*/HDIM, /*tile_d1=*/64, /*stride0=*/D_MODEL,
                /*pad_interval=*/5, /*pad_amount=*/3);       // 256B rows +16B
    const __bf16* vg = Vtg + (size_t)(b * NHEAD + h) * HDIM * SEQ + kt * 64;
    tdm_load_2d((unsigned int)(size_t)&Vt[buf][0][0], vg,
                /*tile_d0=*/64, /*tile_d1=*/HDIM, /*stride0=*/SEQ,
                /*pad_interval=*/4, /*pad_amount=*/3);       // 128B rows +16B
  };

  if (wave == 0) {
    issue_tile(0, 0);
    __builtin_amdgcn_s_wait_tensorcnt(0);
  }
  __syncthreads();

  for (int kt = 0; kt < NT; ++kt) {
    const int cur = kt & 1;
    const int kb0 = kt * 64;
    if (wave == 0 && (kt + 1) < NT) issue_tile(kt + 1, cur ^ 1);   // prefetch

    // Scores: wave computes q rows [wave*16, +16) x 64 keys.
#pragma unroll
    for (int nf = 0; nf < 4; ++nf) {
      v8f s = vzero;
#pragma unroll
      for (int ks = 0; ks < 4; ++ks) {
        const v16bf kfrag = frag_b(&Ks[cur][nf * 16][ks * 32], 136, lane);
        s = __builtin_amdgcn_wmma_f32_16x16x32_bf16(
            false, qa[ks], false, kfrag, (short)0, s, false, false);
      }
      // scale + ReLU; emit f32 attention weights and bf16 P tile.
      const int colL = nf * 16 + ccol;
      float* awp = attn_w +
          ((size_t)((b * NHEAD + h) * SEQ) + q0 + wave * 16 + crow) * SEQ + kb0 + colL;
#pragma unroll
      for (int i = 0; i < 8; ++i) {
        float v = s[i] * scale;
        v = v > 0.0f ? v : 0.0f;
        awp[(size_t)i * SEQ] = v;
        Ps[wave * 16 + crow + i][colL] = (__bf16)v;
      }
    }
    __syncthreads();   // Ps visible to all waves

    // Context accumulate: cacc(16x128) += P(16x64) @ V(64x128).
#pragma unroll
    for (int ks = 0; ks < 2; ++ks) {
      const v16bf pa = frag_a(&Ps[wave * 16][ks * 32], 72, lane);
#pragma unroll
      for (int nf = 0; nf < 8; ++nf) {
        const v16bf vb = frag_b(&Vt[cur][nf * 16][ks * 32], 72, lane);
        cacc[nf] = __builtin_amdgcn_wmma_f32_16x16x32_bf16(
            false, pa, false, vb, (short)0, cacc[nf], false, false);
      }
    }

    if (wave == 0) __builtin_amdgcn_s_wait_tensorcnt(0);   // next tiles landed
    __syncthreads();   // all waves see new tiles; Ps reads done before rewrite
  }

  // Write context in [B,S,D] bf16 layout for the output projection GEMM.
#pragma unroll
  for (int nf = 0; nf < 8; ++nf) {
#pragma unroll
    for (int i = 0; i < 8; ++i) {
      const int row = q0 + wave * 16 + crow + i;
      const int hd  = nf * 16 + ccol;
      ctx[(size_t)(b * SEQ + row) * D_MODEL + h * HDIM + hd] = (__bf16)cacc[nf][i];
    }
  }
}

// ---- Host-side launch -------------------------------------------------------
extern "C" void kernel_launch(void* const* d_in, const int* in_sizes, int n_in,
                              void* d_out, int out_size, void* d_ws, size_t ws_size,
                              hipStream_t stream) {
  const float* query = (const float*)d_in[0];
  const float* keyx  = (const float*)d_in[1];
  const float* value = (const float*)d_in[2];
  const float* Wq = (const float*)d_in[3];
  const float* bq = (const float*)d_in[4];
  const float* Wk = (const float*)d_in[5];
  const float* bk = (const float*)d_in[6];
  const float* Wv = (const float*)d_in[7];
  const float* bv = (const float*)d_in[8];
  const float* Wo = (const float*)d_in[9];
  const float* bo = (const float*)d_in[10];

  float* out    = (float*)d_out;                          // [B,S,D] f32
  float* attn_w = out + (size_t)BATCH * SEQ * D_MODEL;    // [B,H,S,S] f32

  const size_t tok = (size_t)BATCH * SEQ * D_MODEL;       // 4 Mi elements
  __bf16* Qb  = (__bf16*)d_ws;                            // [B*S, D]
  __bf16* Kb  = Qb + tok;                                 // [B*S, D]
  __bf16* Vtg = Kb + tok;                                 // [B,H,HDIM,SEQ]
  __bf16* Cb  = Vtg + tok;                                // [B*S, D]  (32 MiB)

  const int M = BATCH * SEQ;                              // 4096
  dim3 gg(D_MODEL / 128, M / 128);                        // (8, 32)

  gemm_bias_kernel<true, 0><<<gg, 256, 0, stream>>>(query, Wq, bq, Qb, M, D_MODEL, D_MODEL);
  gemm_bias_kernel<true, 0><<<gg, 256, 0, stream>>>(keyx,  Wk, bk, Kb, M, D_MODEL, D_MODEL);
  gemm_bias_kernel<true, 2><<<gg, 256, 0, stream>>>(value, Wv, bv, Vtg, M, D_MODEL, D_MODEL);

  attn_kernel<<<dim3(SEQ / 64, NHEAD, BATCH), 128, 0, stream>>>(Qb, Kb, Vtg, attn_w, Cb);

  gemm_bias_kernel<false, 1><<<gg, 256, 0, stream>>>(Cb, Wo, bo, out, M, D_MODEL, D_MODEL);
}
